// SparseProcessor_28467043238117
// MI455X (gfx1250) — matrix operand order repaired and checked by reading, not compiled, not run on hardware
//
#include <hip/hip_runtime.h>
#include <hip/hip_bf16.h>
#include <stdint.h>

#define N_NODES   40962
#define D_DIM     512
#define N_EDGES   655392
#define N_BLOCKS  4
#define N_HEADS   4
#define HEAD_DIM  128
#define C_DIM     16
#define N_FREQ    32

typedef __attribute__((ext_vector_type(16))) __bf16 v16bf;
typedef __attribute__((ext_vector_type(8)))  float  v8f;

union BfPack { v16bf v; uint4 u[2]; };

__device__ __forceinline__ unsigned short f2bf(float f) {
  unsigned u = __float_as_uint(f);
  u += 0x7FFFu + ((u >> 16) & 1u);           // round-to-nearest-even
  return (unsigned short)(u >> 16);
}

// monotone unsigned mapping of float for integer atomicMax
__device__ __forceinline__ unsigned ford(float f) {
  unsigned u = __float_as_uint(f);
  return (u & 0x80000000u) ? ~u : (u | 0x80000000u);
}
__device__ __forceinline__ float dford(unsigned u) {
  u = (u & 0x80000000u) ? (u & 0x7FFFFFFFu) : ~u;
  return __uint_as_float(u);
}

// ---------------------------------------------------------------------------
// Fourier noise embedding + 2-layer SiLU MLP -> cond (N,16)
// ---------------------------------------------------------------------------
__global__ void cond_kernel(const float* __restrict__ noise,
                            const float* __restrict__ w1, const float* __restrict__ b1,
                            const float* __restrict__ w2, const float* __restrict__ b2,
                            float* __restrict__ cond) {
  int n = blockIdx.x * blockDim.x + threadIdx.x;
  if (n >= N_NODES) return;
  float nl = noise[n];
  const float w0 = 2.0f * 3.14159265358979323846f / 16.0f;
  float emb[2 * N_FREQ];
  #pragma unroll
  for (int f = 0; f < N_FREQ; ++f) {
    float ph = nl * (float)(f + 1) * w0;
    emb[f] = __sinf(ph);
    emb[N_FREQ + f] = __cosf(ph);
  }
  float h[C_DIM];
  #pragma unroll
  for (int c = 0; c < C_DIM; ++c) {
    float a = b1[c];
    for (int j = 0; j < 2 * N_FREQ; ++j) a += emb[j] * w1[j * C_DIM + c];
    h[c] = a / (1.0f + __expf(-a));          // SiLU
  }
  #pragma unroll
  for (int c = 0; c < C_DIM; ++c) {
    float a = b2[c];
    for (int j = 0; j < C_DIM; ++j) a += h[j] * w2[j * C_DIM + c];
    cond[(size_t)n * C_DIM + c] = a;
  }
}

// ---------------------------------------------------------------------------
// Conditional LayerNorm + FiLM -> bf16 activations (wave per node)
// ---------------------------------------------------------------------------
__global__ void __launch_bounds__(256) cln_kernel(
    const float* __restrict__ x, const float* __restrict__ cond,
    const float* __restrict__ sw, const float* __restrict__ sb,
    const float* __restrict__ bw, const float* __restrict__ bb,
    unsigned short* __restrict__ ybf) {
  int n = blockIdx.x * 8 + (threadIdx.x >> 5);
  int lane = threadIdx.x & 31;
  if (n >= N_NODES) return;
  const float* xr = x + (size_t)n * D_DIM;
  float vals[16];
  float s = 0.f, sq = 0.f;
  #pragma unroll
  for (int j = 0; j < 16; ++j) {
    float t = xr[lane + 32 * j];
    vals[j] = t; s += t; sq += t * t;
  }
  for (int off = 16; off >= 1; off >>= 1) {
    s  += __shfl_xor(s,  off, 32);
    sq += __shfl_xor(sq, off, 32);
  }
  float mu   = s  * (1.0f / 512.0f);
  float var  = sq * (1.0f / 512.0f) - mu * mu;
  float rinv = rsqrtf(var + 1e-5f);
  float c[C_DIM];
  const float* cr = cond + (size_t)n * C_DIM;
  #pragma unroll
  for (int cc = 0; cc < C_DIM; ++cc) c[cc] = cr[cc];
  #pragma unroll
  for (int j = 0; j < 16; ++j) {
    int d = lane + 32 * j;
    float sc = sb[d], bi = bb[d];
    #pragma unroll
    for (int cc = 0; cc < C_DIM; ++cc) {
      sc += c[cc] * sw[cc * D_DIM + d];
      bi += c[cc] * bw[cc * D_DIM + d];
    }
    float xn = (vals[j] - mu) * rinv;
    ybf[(size_t)n * D_DIM + d] = f2bf(xn * (1.0f + sc) + bi);
  }
}

// ---------------------------------------------------------------------------
// Tiled WMMA GEMM: out(rows,512) = A(rows,512)bf16 @ Wt(512,512)bf16^T + bias
// Wt stored K-contiguous per output column (Wt[n][k]) -> contiguous fragments.
// One wave -> 64x64 output tile (4x4 WMMA accumulators): per K-step,
// 8 x (2xb128) fragment loads feed 16 independent WMMAs (AI ~32 FLOP/B).
// 8 waves/WG -> 64 rows x 512 cols; A rows shared across waves via WGP$.
// Fused: optional residual add, ReLU, f32 or bf16 store.
// ---------------------------------------------------------------------------
__global__ void __launch_bounds__(256) gemm512_wmma(
    const unsigned short* __restrict__ A,
    const unsigned short* __restrict__ Wt,
    const float* __restrict__ bias,
    const float* __restrict__ resid,      // nullable
    float* __restrict__ outF,             // nullable (else outBf)
    unsigned short* __restrict__ outBf,   // nullable
    int relu, int rows) {
  const int lane = threadIdx.x & 31;
  const int wave = threadIdx.x >> 5;
  const int mBase = blockIdx.x * 64;
  const int colBase = wave * 64;
  const int l15 = lane & 15;
  const int hi = lane >> 4;                              // 0 or 1

  // A fragment bases: lane<16 holds K 0..7 / 16..23, lane>=16 K 8..15 / 24..31
  const unsigned short* aRow[4];
  #pragma unroll
  for (int i = 0; i < 4; ++i) {
    int r = mBase + i * 16 + l15;
    if (r >= rows) r = rows - 1;                         // clamp pad rows
    aRow[i] = A + (size_t)r * D_DIM + hi * 8;
  }
  // B fragment base: lane<16 holds K 0..7 / 16..23, lane>=16 K 8..15 / 24..31
  const unsigned short* bCol = Wt + (size_t)(colBase + l15) * D_DIM + hi * 8;

  v8f zero = {0.f, 0.f, 0.f, 0.f, 0.f, 0.f, 0.f, 0.f};
  v8f acc[4][4];
  #pragma unroll
  for (int i = 0; i < 4; ++i)
    #pragma unroll
    for (int j = 0; j < 4; ++j) acc[i][j] = zero;

  for (int kk = 0; kk < D_DIM; kk += 32) {
    BfPack a[4], b[4];
    #pragma unroll
    for (int i = 0; i < 4; ++i) {
      a[i].u[0] = *(const uint4*)(aRow[i] + kk);
      a[i].u[1] = *(const uint4*)(aRow[i] + kk + 16);
    }
    #pragma unroll
    for (int j = 0; j < 4; ++j) {
      const unsigned short* bp = bCol + (size_t)j * 16 * D_DIM + kk;
      b[j].u[0] = *(const uint4*)(bp);
      b[j].u[1] = *(const uint4*)(bp + 16);
    }
    #pragma unroll
    for (int i = 0; i < 4; ++i)
      #pragma unroll
      for (int j = 0; j < 4; ++j)
        acc[i][j] = __builtin_amdgcn_wmma_f32_16x16x32_bf16(
            false, a[i].v, false, b[j].v, (short)0, acc[i][j], false, false);
  }

  #pragma unroll
  for (int j = 0; j < 4; ++j) {
    const int col = colBase + j * 16 + l15;
    const float bv = bias[col];
    #pragma unroll
    for (int i = 0; i < 4; ++i) {
      #pragma unroll
      for (int e = 0; e < 8; ++e) {
        const int row = mBase + i * 16 + hi * 8 + e;     // C/D layout
        if (row < rows) {
          const size_t idx = (size_t)row * D_DIM + col;
          float v = acc[i][j][e] + bv;
          if (resid) v += resid[idx];
          if (relu)  v = fmaxf(v, 0.f);
          if (outF)  outF[idx] = v;
          else       outBf[idx] = f2bf(v);
        }
      }
    }
  }
}

// ---------------------------------------------------------------------------
// Weight convert + transpose: W[b][k][n] f32 -> Wt[b][n][k] bf16
// ---------------------------------------------------------------------------
__global__ void wconv_kernel(const float* __restrict__ w,
                             unsigned short* __restrict__ wt, int total) {
  int i = blockIdx.x * blockDim.x + threadIdx.x;
  if (i >= total) return;
  int b = i >> 18;                 // /(512*512)
  int r = i & 262143;
  int n = r >> 9, k = r & 511;
  wt[i] = f2bf(w[(size_t)b * 262144 + (size_t)k * 512 + n]);
}

// ---------------------------------------------------------------------------
// Attention scratch init
// ---------------------------------------------------------------------------
__global__ void attn_init_kernel(unsigned* __restrict__ smax,
                                 float* __restrict__ denom,
                                 float* __restrict__ acc) {
  int i = blockIdx.x * blockDim.x + threadIdx.x;
  if (i < N_NODES * N_HEADS) { smax[i] = 0u; denom[i] = 0.f; }
  if (i < N_NODES * D_DIM)   acc[i] = 0.f;
}

// ---------------------------------------------------------------------------
// Pass 1: per-edge per-head scores + segment max (wave per edge)
// ---------------------------------------------------------------------------
__global__ void __launch_bounds__(256) edge_scores_kernel(
    const float* __restrict__ q, const float* __restrict__ k,
    const int* __restrict__ src, const int* __restrict__ dst,
    float* __restrict__ scores, unsigned* __restrict__ smax) {
  int e = blockIdx.x * 8 + (threadIdx.x >> 5);
  int lane = threadIdx.x & 31;
  if (e >= N_EDGES) return;
  int s = src[e], d = dst[e];
  const float scale = 0.08838834764831845f;              // 128^-0.5
  #pragma unroll
  for (int h = 0; h < N_HEADS; ++h) {
    float4 qa = *(const float4*)(q + (size_t)s * D_DIM + h * HEAD_DIM + lane * 4);
    float4 ka = *(const float4*)(k + (size_t)d * D_DIM + h * HEAD_DIM + lane * 4);
    float p = qa.x * ka.x + qa.y * ka.y + qa.z * ka.z + qa.w * ka.w;
    for (int off = 16; off >= 1; off >>= 1) p += __shfl_xor(p, off, 32);
    if (lane == 0) {
      float sc = p * scale;
      scores[(size_t)e * N_HEADS + h] = sc;
      atomicMax(&smax[(size_t)s * N_HEADS + h], ford(sc));
    }
  }
}

// ---------------------------------------------------------------------------
// Pass 2: exp, denom accumulate, weighted scatter of v[dst] into acc[src]
// ---------------------------------------------------------------------------
__global__ void __launch_bounds__(256) edge_scatter_kernel(
    const float* __restrict__ scores, const unsigned* __restrict__ smax,
    const float* __restrict__ v,
    const int* __restrict__ src, const int* __restrict__ dst,
    float* __restrict__ denom, float* __restrict__ acc) {
  int e = blockIdx.x * 8 + (threadIdx.x >> 5);
  int lane = threadIdx.x & 31;
  if (e >= N_EDGES) return;
  int s = src[e], d = dst[e];
  #pragma unroll
  for (int h = 0; h < N_HEADS; ++h) {
    float sc = scores[(size_t)e * N_HEADS + h];
    float m  = dford(smax[(size_t)s * N_HEADS + h]);
    float ex = __expf(sc - m);
    if (lane == 0) atomicAdd(&denom[(size_t)s * N_HEADS + h], ex);
    float4 va = *(const float4*)(v + (size_t)d * D_DIM + h * HEAD_DIM + lane * 4);
    float* ap = acc + (size_t)s * D_DIM + h * HEAD_DIM + lane * 4;
    atomicAdd(ap + 0, ex * va.x);
    atomicAdd(ap + 1, ex * va.y);
    atomicAdd(ap + 2, ex * va.z);
    atomicAdd(ap + 3, ex * va.w);
  }
}

// ---------------------------------------------------------------------------
// Softmax normalize + bf16 convert (input to wo projection)
// ---------------------------------------------------------------------------
__global__ void attn_norm_kernel(const float* __restrict__ acc,
                                 const float* __restrict__ denom,
                                 unsigned short* __restrict__ outbf) {
  int i = blockIdx.x * blockDim.x + threadIdx.x;
  if (i >= N_NODES * D_DIM) return;
  int n = i >> 9;
  int h = (i >> 7) & 3;
  outbf[i] = f2bf(acc[i] / denom[(size_t)n * N_HEADS + h]);
}

// ---------------------------------------------------------------------------
extern "C" void kernel_launch(void* const* d_in, const int* in_sizes, int n_in,
                              void* d_out, int out_size, void* d_ws, size_t ws_size,
                              hipStream_t stream) {
  (void)in_sizes; (void)n_in; (void)out_size; (void)ws_size;

  const float* x_in   = (const float*)d_in[0];
  const int*   eidx   = (const int*)d_in[1];
  const float* noise  = (const float*)d_in[2];
  const float* f_w1   = (const float*)d_in[3];
  const float* f_b1   = (const float*)d_in[4];
  const float* f_w2   = (const float*)d_in[5];
  const float* f_b2   = (const float*)d_in[6];
  const float* s1_w   = (const float*)d_in[7];
  const float* s1_b   = (const float*)d_in[8];
  const float* b1_w   = (const float*)d_in[9];
  const float* b1_b   = (const float*)d_in[10];
  const float* wq     = (const float*)d_in[11];
  const float* bq     = (const float*)d_in[12];
  const float* wk     = (const float*)d_in[13];
  const float* bk     = (const float*)d_in[14];
  const float* wv     = (const float*)d_in[15];
  const float* bv     = (const float*)d_in[16];
  const float* wo     = (const float*)d_in[17];
  const float* bo     = (const float*)d_in[18];
  const float* s2_w   = (const float*)d_in[19];
  const float* s2_b   = (const float*)d_in[20];
  const float* b2_w   = (const float*)d_in[21];
  const float* b2_b   = (const float*)d_in[22];
  const float* m_w1   = (const float*)d_in[23];
  const float* m_b1   = (const float*)d_in[24];
  const float* m_w2   = (const float*)d_in[25];
  const float* m_b2   = (const float*)d_in[26];

  const int* src = eidx;
  const int* dst = eidx + N_EDGES;

  // ---- workspace bump allocator (256B aligned chunks) ----
  char* p = (char*)d_ws;
  auto alloc = [&](size_t bytes) -> void* {
    void* r = (void*)p;
    p += (bytes + 255) & ~(size_t)255;
    return r;
  };
  const size_t WBYTES = (size_t)N_BLOCKS * D_DIM * D_DIM * sizeof(unsigned short);
  unsigned short* Wq_bf = (unsigned short*)alloc(WBYTES);
  unsigned short* Wk_bf = (unsigned short*)alloc(WBYTES);
  unsigned short* Wv_bf = (unsigned short*)alloc(WBYTES);
  unsigned short* Wo_bf = (unsigned short*)alloc(WBYTES);
  unsigned short* W1_bf = (unsigned short*)alloc(WBYTES);
  unsigned short* W2_bf = (unsigned short*)alloc(WBYTES);
  float*          cond  = (float*)alloc((size_t)N_NODES * C_DIM * 4);
  unsigned short* y_bf  = (unsigned short*)alloc((size_t)N_NODES * D_DIM * 2);
  unsigned short* a_bf  = (unsigned short*)alloc((size_t)N_NODES * D_DIM * 2);
  float*          qb    = (float*)alloc((size_t)N_NODES * D_DIM * 4);
  float*          kb    = (float*)alloc((size_t)N_NODES * D_DIM * 4);
  float*          vb    = (float*)alloc((size_t)N_NODES * D_DIM * 4);
  float*          accb  = (float*)alloc((size_t)N_NODES * D_DIM * 4);
  float*          sc    = (float*)alloc((size_t)N_EDGES * N_HEADS * 4);
  unsigned*       smax  = (unsigned*)alloc((size_t)N_NODES * N_HEADS * 4);
  float*          denom = (float*)alloc((size_t)N_NODES * N_HEADS * 4);

  float* x = (float*)d_out;   // residual stream lives in d_out

  // residual stream init
  hipMemcpyAsync(x, x_in, (size_t)N_NODES * D_DIM * sizeof(float),
                 hipMemcpyDeviceToDevice, stream);

  // weight conversion (all blocks at once, per tensor)
  const int WTOT = N_BLOCKS * D_DIM * D_DIM;
  const int WGRD = (WTOT + 255) / 256;
  wconv_kernel<<<WGRD, 256, 0, stream>>>(wq,   Wq_bf, WTOT);
  wconv_kernel<<<WGRD, 256, 0, stream>>>(wk,   Wk_bf, WTOT);
  wconv_kernel<<<WGRD, 256, 0, stream>>>(wv,   Wv_bf, WTOT);
  wconv_kernel<<<WGRD, 256, 0, stream>>>(wo,   Wo_bf, WTOT);
  wconv_kernel<<<WGRD, 256, 0, stream>>>(m_w1, W1_bf, WTOT);
  wconv_kernel<<<WGRD, 256, 0, stream>>>(m_w2, W2_bf, WTOT);

  // noise conditioning
  cond_kernel<<<(N_NODES + 255) / 256, 256, 0, stream>>>(noise, f_w1, f_b1, f_w2, f_b2, cond);

  const int MT     = (N_NODES + 63) / 64;        // GEMM row tiles (64 rows/WG)
  const int NWG    = (N_NODES + 7) / 8;          // wave-per-node grids
  const int EWG    = (N_EDGES + 7) / 8;          // wave-per-edge grids
  const int ELGRD  = (N_NODES * D_DIM + 255) / 256;
  const size_t WOFF = (size_t)D_DIM * D_DIM;

  for (int i = 0; i < N_BLOCKS; ++i) {
    // --- CLN1 + FiLM ---
    cln_kernel<<<NWG, 256, 0, stream>>>(
        x, cond, s1_w + (size_t)i * C_DIM * D_DIM, s1_b + (size_t)i * D_DIM,
        b1_w + (size_t)i * C_DIM * D_DIM, b1_b + (size_t)i * D_DIM, y_bf);

    // --- Q,K,V projections (WMMA) ---
    gemm512_wmma<<<MT, 256, 0, stream>>>(y_bf, Wq_bf + (size_t)i * WOFF,
        bq + (size_t)i * D_DIM, nullptr, qb, nullptr, 0, N_NODES);
    gemm512_wmma<<<MT, 256, 0, stream>>>(y_bf, Wk_bf + (size_t)i * WOFF,
        bk + (size_t)i * D_DIM, nullptr, kb, nullptr, 0, N_NODES);
    gemm512_wmma<<<MT, 256, 0, stream>>>(y_bf, Wv_bf + (size_t)i * WOFF,
        bv + (size_t)i * D_DIM, nullptr, vb, nullptr, 0, N_NODES);

    // --- sparse softmax attention ---
    attn_init_kernel<<<ELGRD, 256, 0, stream>>>(smax, denom, accb);
    edge_scores_kernel<<<EWG, 256, 0, stream>>>(qb, kb, src, dst, sc, smax);
    edge_scatter_kernel<<<EWG, 256, 0, stream>>>(sc, smax, vb, src, dst, denom, accb);
    attn_norm_kernel<<<ELGRD, 256, 0, stream>>>(accb, denom, a_bf);

    // --- output projection + residual (WMMA, in-place on x) ---
    gemm512_wmma<<<MT, 256, 0, stream>>>(a_bf, Wo_bf + (size_t)i * WOFF,
        bo + (size_t)i * D_DIM, x, x, nullptr, 0, N_NODES);

    // --- CLN2 + FiLM ---
    cln_kernel<<<NWG, 256, 0, stream>>>(
        x, cond, s2_w + (size_t)i * C_DIM * D_DIM, s2_b + (size_t)i * D_DIM,
        b2_w + (size_t)i * C_DIM * D_DIM, b2_b + (size_t)i * D_DIM, y_bf);

    // --- FFN: ReLU GEMM -> bf16 hidden, then GEMM + residual (WMMA) ---
    gemm512_wmma<<<MT, 256, 0, stream>>>(y_bf, W1_bf + (size_t)i * WOFF,
        m_b1 + (size_t)i * D_DIM, nullptr, nullptr, a_bf, 1, N_NODES);
    gemm512_wmma<<<MT, 256, 0, stream>>>(a_bf, W2_bf + (size_t)i * WOFF,
        m_b2 + (size_t)i * D_DIM, x, x, nullptr, 0, N_NODES);
  }
}
